// GraphEncoder_59536836657700
// MI455X (gfx1250) — matrix operand order repaired and checked by reading, not compiled, not run on hardware
//
#include <hip/hip_runtime.h>
#include <hip/hip_bf16.h>

typedef __attribute__((ext_vector_type(16))) __bf16 v16bf;
typedef __attribute__((ext_vector_type(8)))  float  v8f;

// ---------------- WMMA A-fragment loader (16x16x32 bf16, contiguous k = 16*half + s order) ----------------
// A: 16x32 (MxK). Lane L(0..15): row M=L, K = k0+[0..15].
// Lane L+16: row M=L, K = k0+[16..31].
__device__ __forceinline__ v16bf load_a_frag(const float* __restrict__ A, int lda,
                                             int row, int k0, int half, float scale) {
    const float* rowp = A + (size_t)row * lda + k0 + half * 16;
    v16bf a;
#pragma unroll
    for (int jj = 0; jj < 4; ++jj) {
        float2 lo = *(const float2*)(rowp + 2 * jj);
        float2 hi = *(const float2*)(rowp + 8 + 2 * jj);
        a[2 * jj]     = (__bf16)(lo.x * scale);
        a[2 * jj + 1] = (__bf16)(lo.y * scale);
        a[8 + 2 * jj] = (__bf16)(hi.x * scale);
        a[9 + 2 * jj] = (__bf16)(hi.y * scale);
    }
    return a;
}

// ---------------- Weight prepack: fp32 [K x Nout] -> bf16 B-fragments ----------------
// Fragment (kt, fn): lane l holds column n = fn*16 + (l&15); lanes 0-15 take K = kt*32+[0..15],
// lanes 16-31 take K = kt*32+16+[0..15]. Stored per-lane contiguous: 16 bf16 = 32 bytes.
// Packed index: frag = fn*kTiles + kt ; offset = (frag*32 + lane)*16 elements.
__global__ void prepack_w(const float* __restrict__ W, __bf16* __restrict__ Wp,
                          int kTiles, int Nout, int total) {
    int t = blockIdx.x * blockDim.x + threadIdx.x;
    if (t >= total) return;
    int lane = t & 31;
    int frag = t >> 5;
    int fn = frag / kTiles;
    int kt = frag - fn * kTiles;
    int halfL = lane >> 4;
    int n = fn * 16 + (lane & 15);
    const float* colp = W + (size_t)(kt * 32 + halfL * 16) * Nout + n;
    __bf16* o = Wp + ((size_t)frag * 32 + lane) * 16;
#pragma unroll
    for (int i = 0; i < 16; ++i) o[i] = (__bf16)colp[(size_t)i * Nout];
}

__device__ __forceinline__ v16bf load_b_packed(const __bf16* __restrict__ Wp,
                                               int kTiles, int kt, int fn, int lane) {
    return *(const v16bf*)(Wp + ((size_t)(fn * kTiles + kt) * 32 + lane) * 16);
}

// ---------------- Fused GEMM: Out = act( A1@W1 + (A2/deg)@W2 + bias ) ----------------
// One wave computes a 16x64 strip (4 accumulators); A fragment reused across 4 WMMAs.
// Requires M % 16 == 0, Nout % 64 == 0. blockDim = (32,4): 4 waves = 4 consecutive mTiles.
__global__ void wmma_gemm_fused(const float* __restrict__ A1, const __bf16* __restrict__ W1p,
                                const float* __restrict__ A2, const __bf16* __restrict__ W2p,
                                const float* __restrict__ deg, const float* __restrict__ bias,
                                float* __restrict__ Out,
                                int Mtiles, int K, int lda1, int lda2, int ldo, int doAct) {
    const int lane  = threadIdx.x & 31;
    const int half  = lane >> 4;
    const int mTile = blockIdx.x * 4 + threadIdx.y;
    if (mTile >= Mtiles) return;              // wave-uniform guard (EXEC stays all-ones)
    const int row0 = mTile * 16;
    const int fn0  = blockIdx.y * 4;          // group of 4 n-tiles (64 columns)
    const int kTiles = K >> 5;
    const int rowA = row0 + (lane & 15);

    float sc2 = 1.0f;
    if (A2 != nullptr && deg != nullptr) sc2 = 1.0f / fmaxf(deg[rowA], 1.0f);

    v8f acc0 = {0.f,0.f,0.f,0.f,0.f,0.f,0.f,0.f};
    v8f acc1 = acc0, acc2 = acc0, acc3 = acc0;

    for (int kt = 0; kt < kTiles; ++kt) {
        const int k0 = kt * 32;
        v16bf a = load_a_frag(A1, lda1, rowA, k0, half, 1.0f);
        acc0 = __builtin_amdgcn_wmma_f32_16x16x32_bf16(false, a, false,
                 load_b_packed(W1p, kTiles, kt, fn0 + 0, lane), (short)0, acc0, false, false);
        acc1 = __builtin_amdgcn_wmma_f32_16x16x32_bf16(false, a, false,
                 load_b_packed(W1p, kTiles, kt, fn0 + 1, lane), (short)0, acc1, false, false);
        acc2 = __builtin_amdgcn_wmma_f32_16x16x32_bf16(false, a, false,
                 load_b_packed(W1p, kTiles, kt, fn0 + 2, lane), (short)0, acc2, false, false);
        acc3 = __builtin_amdgcn_wmma_f32_16x16x32_bf16(false, a, false,
                 load_b_packed(W1p, kTiles, kt, fn0 + 3, lane), (short)0, acc3, false, false);
        if (A2 != nullptr) {
            v16bf a2 = load_a_frag(A2, lda2, rowA, k0, half, sc2);
            acc0 = __builtin_amdgcn_wmma_f32_16x16x32_bf16(false, a2, false,
                     load_b_packed(W2p, kTiles, kt, fn0 + 0, lane), (short)0, acc0, false, false);
            acc1 = __builtin_amdgcn_wmma_f32_16x16x32_bf16(false, a2, false,
                     load_b_packed(W2p, kTiles, kt, fn0 + 1, lane), (short)0, acc1, false, false);
            acc2 = __builtin_amdgcn_wmma_f32_16x16x32_bf16(false, a2, false,
                     load_b_packed(W2p, kTiles, kt, fn0 + 2, lane), (short)0, acc2, false, false);
            acc3 = __builtin_amdgcn_wmma_f32_16x16x32_bf16(false, a2, false,
                     load_b_packed(W2p, kTiles, kt, fn0 + 3, lane), (short)0, acc3, false, false);
        }
    }

    // Epilogue: C/D layout: vgpr r -> row = row0 + r + 8*half, col = tile_n0 + (lane&15)
    v8f accs[4] = {acc0, acc1, acc2, acc3};
#pragma unroll
    for (int j = 0; j < 4; ++j) {
        const int col = (fn0 + j) * 16 + (lane & 15);
        const float bv = (bias != nullptr) ? bias[col] : 0.0f;
#pragma unroll
        for (int r = 0; r < 8; ++r) {
            const int row = row0 + r + half * 8;
            float v = accs[j][r] + bv;
            if (doAct) v = (v >= 0.0f) ? v : 0.01f * v;
            Out[(size_t)row * ldo + col] = v;
        }
    }
}

// ---------------- Node projection: x = pos @ w_proj + b_proj ----------------
__global__ void proj_kernel(const float* __restrict__ pos, const float* __restrict__ w,
                            const float* __restrict__ b, float* __restrict__ x, int Nn) {
    int t = blockIdx.x * blockDim.x + threadIdx.x;
    if (t >= Nn * 64) return;
    int i = t >> 6, j = t & 63;
    float p0 = pos[i * 3 + 0], p1 = pos[i * 3 + 1], p2 = pos[i * 3 + 2];
    x[t] = p0 * w[j] + p1 * w[64 + j] + p2 * w[128 + j] + b[j];
}

// ---------------- In-degree count ----------------
__global__ void deg_kernel(const int* __restrict__ dstIdx, float* __restrict__ deg, int E) {
    int e = blockIdx.x * blockDim.x + threadIdx.x;
    if (e < E) atomicAdd(&deg[dstIdx[e]], 1.0f);
}

// ---------------- Edge scatter-add: Agg[dst] += H[src] ----------------
__global__ void scatter_add_kernel(const float* __restrict__ H, int ldh,
                                   const int* __restrict__ srcIdx, const int* __restrict__ dstIdx,
                                   float* __restrict__ Agg, int ldagg, int F4, long long total) {
    long long t = (long long)blockIdx.x * blockDim.x + threadIdx.x;
    if (t >= total) return;
    int e = (int)(t / F4);
    int c = (int)(t % F4) * 4;
    int s = srcIdx[e], d = dstIdx[e];
    float4 v = *(const float4*)(H + (size_t)s * ldh + c);
    float* o = Agg + (size_t)d * ldagg + c;
    atomicAdd(o + 0, v.x);
    atomicAdd(o + 1, v.y);
    atomicAdd(o + 2, v.z);
    atomicAdd(o + 3, v.w);
}

// ---------------- Pooling: monotonic-uint atomicMax + sum + count ----------------
__device__ __forceinline__ unsigned mapf(float f) {
    unsigned b = __float_as_uint(f);
    return (b & 0x80000000u) ? ~b : (b | 0x80000000u);
}

__global__ void pool_kernel(const float* __restrict__ nf, const int* __restrict__ batch,
                            unsigned* __restrict__ gmaxU, float* __restrict__ gsum,
                            float* __restrict__ gcnt, int Nn) {
    int t = blockIdx.x * blockDim.x + threadIdx.x;
    if (t >= Nn * 64) return;
    int i = t >> 6;
    int c = (t & 63) * 4;
    int g = batch[i];
    float4 v = *(const float4*)(nf + (size_t)i * 256 + c);
    unsigned* mo = gmaxU + (size_t)g * 256 + c;
    float*    so = gsum  + (size_t)g * 256 + c;
    atomicMax(mo + 0, mapf(v.x)); atomicAdd(so + 0, v.x);
    atomicMax(mo + 1, mapf(v.y)); atomicAdd(so + 1, v.y);
    atomicMax(mo + 2, mapf(v.z)); atomicAdd(so + 2, v.z);
    atomicMax(mo + 3, mapf(v.w)); atomicAdd(so + 3, v.w);
    if ((t & 63) == 0) atomicAdd(&gcnt[g], 1.0f);
}

__global__ void pool_finalize(const unsigned* __restrict__ gmaxU, const float* __restrict__ gsum,
                              const float* __restrict__ gcnt, float* __restrict__ g, int G) {
    int t = blockIdx.x * blockDim.x + threadIdx.x;
    if (t >= G * 512) return;
    int gi = t >> 9, f = t & 511;
    float v;
    if (f < 256) {
        unsigned u = gmaxU[gi * 256 + f];
        unsigned b = (u & 0x80000000u) ? (u ^ 0x80000000u) : ~u;
        v = __uint_as_float(b);
    } else {
        v = gsum[gi * 256 + (f - 256)] / fmaxf(gcnt[gi], 1.0f);
    }
    g[t] = v;
}

// ---------------- Host orchestration ----------------
extern "C" void kernel_launch(void* const* d_in, const int* in_sizes, int n_in,
                              void* d_out, int out_size, void* d_ws, size_t ws_size,
                              hipStream_t stream) {
    const int N = in_sizes[0] / 3;
    const int E = in_sizes[1] / 2;
    const int G = out_size / 256;

    const float* pos  = (const float*)d_in[0];
    const int*   ei   = (const int*)d_in[1];
    const int*   bat  = (const int*)d_in[2];
    const float* wpr  = (const float*)d_in[3];
    const float* bpr  = (const float*)d_in[4];
    const float* ws0  = (const float*)d_in[5];
    const float* wn0  = (const float*)d_in[6];
    const float* bb0  = (const float*)d_in[7];
    const float* ws1  = (const float*)d_in[8];
    const float* wn1  = (const float*)d_in[9];
    const float* bb1  = (const float*)d_in[10];
    const float* ws2  = (const float*)d_in[11];
    const float* wn2  = (const float*)d_in[12];
    const float* bb2  = (const float*)d_in[13];
    const float* ws3  = (const float*)d_in[14];
    const float* wn3  = (const float*)d_in[15];
    const float* bb3  = (const float*)d_in[16];
    const float* wd1  = (const float*)d_in[17];
    const float* bd1  = (const float*)d_in[18];
    const float* wd2  = (const float*)d_in[19];
    const float* bd2  = (const float*)d_in[20];

    const int* srcI = ei;
    const int* dstI = ei + E;

    // Workspace partition (256B aligned)
    size_t off = 0;
    char* base = (char*)d_ws;
    auto take = [&](size_t bytes) -> void* {
        void* p = base + off;
        off += (bytes + 255) & ~(size_t)255;
        return p;
    };
    float*    x      = (float*)take((size_t)N * 64 * 4);
    float*    hcat   = (float*)take((size_t)N * 256 * 4);
    float*    agg64  = (float*)take((size_t)N * 64 * 4);
    float*    agg256 = (float*)take((size_t)N * 256 * 4);
    float*    nf     = (float*)take((size_t)N * 256 * 4);
    float*    deg    = (float*)take((size_t)N * 4);
    unsigned* gmaxU  = (unsigned*)take((size_t)G * 256 * 4);
    float*    gsum   = (float*)take((size_t)G * 256 * 4);
    float*    gcnt   = (float*)take((size_t)G * 4);
    float*    gvec   = (float*)take((size_t)G * 512 * 4);
    float*    gtmp   = (float*)take((size_t)G * 512 * 4);
    __bf16*   ws0p   = (__bf16*)take((size_t)64 * 64 * 2);
    __bf16*   wn0p   = (__bf16*)take((size_t)64 * 64 * 2);
    __bf16*   ws1p   = (__bf16*)take((size_t)64 * 64 * 2);
    __bf16*   wn1p   = (__bf16*)take((size_t)64 * 64 * 2);
    __bf16*   ws2p   = (__bf16*)take((size_t)64 * 128 * 2);
    __bf16*   wn2p   = (__bf16*)take((size_t)64 * 128 * 2);
    __bf16*   ws3p   = (__bf16*)take((size_t)256 * 256 * 2);
    __bf16*   wn3p   = (__bf16*)take((size_t)256 * 256 * 2);
    __bf16*   wd1p   = (__bf16*)take((size_t)512 * 512 * 2);
    __bf16*   wd2p   = (__bf16*)take((size_t)512 * 256 * 2);
    (void)ws_size; (void)n_in;

    // weight prepack (cheap; re-run every call, fully deterministic)
    auto pack = [&](const float* W, __bf16* Wp, int K, int Nout) {
        int total = (K >> 5) * (Nout >> 4) * 32;
        prepack_w<<<(total + 255) / 256, 256, 0, stream>>>(W, Wp, K >> 5, Nout, total);
    };
    pack(ws0, ws0p, 64, 64);   pack(wn0, wn0p, 64, 64);
    pack(ws1, ws1p, 64, 64);   pack(wn1, wn1p, 64, 64);
    pack(ws2, ws2p, 64, 128);  pack(wn2, wn2p, 64, 128);
    pack(ws3, ws3p, 256, 256); pack(wn3, wn3p, 256, 256);
    pack(wd1, wd1p, 512, 512); pack(wd2, wd2p, 512, 256);

    const dim3 gblk(32, 4);        // 4 waves/block = 4 consecutive mTiles
    const int  mT  = N / 16;       // N = 50000 -> 3125
    const int  mTg = G / 16;       // G = 64 -> 4
    const int  gxN = (mT + 3) / 4;
    const int  gxG = (mTg + 3) / 4;

    // degree counts
    hipMemsetAsync(deg, 0, (size_t)N * 4, stream);
    deg_kernel<<<(E + 255) / 256, 256, 0, stream>>>(dstI, deg, E);

    // projection
    proj_kernel<<<((size_t)N * 64 + 255) / 256, 256, 0, stream>>>(pos, wpr, bpr, x, N);

    // layer 0: x[N,64] -> h0 = hcat[:,0:64]
    hipMemsetAsync(agg64, 0, (size_t)N * 64 * 4, stream);
    {
        long long tot = (long long)E * 16;
        scatter_add_kernel<<<(tot + 255) / 256, 256, 0, stream>>>(x, 64, srcI, dstI, agg64, 64, 16, tot);
    }
    wmma_gemm_fused<<<dim3(gxN, 64 / 64), gblk, 0, stream>>>(
        x, ws0p, agg64, wn0p, deg, bb0, hcat + 0, mT, 64, 64, 64, 256, 1);

    // layer 1: h0 -> h1 = hcat[:,64:128]
    hipMemsetAsync(agg64, 0, (size_t)N * 64 * 4, stream);
    {
        long long tot = (long long)E * 16;
        scatter_add_kernel<<<(tot + 255) / 256, 256, 0, stream>>>(hcat + 0, 256, srcI, dstI, agg64, 64, 16, tot);
    }
    wmma_gemm_fused<<<dim3(gxN, 64 / 64), gblk, 0, stream>>>(
        hcat + 0, ws1p, agg64, wn1p, deg, bb1, hcat + 64, mT, 64, 256, 64, 256, 1);

    // layer 2: h1 -> h2 = hcat[:,128:256] (64 -> 128)
    hipMemsetAsync(agg64, 0, (size_t)N * 64 * 4, stream);
    {
        long long tot = (long long)E * 16;
        scatter_add_kernel<<<(tot + 255) / 256, 256, 0, stream>>>(hcat + 64, 256, srcI, dstI, agg64, 64, 16, tot);
    }
    wmma_gemm_fused<<<dim3(gxN, 128 / 64), gblk, 0, stream>>>(
        hcat + 64, ws2p, agg64, wn2p, deg, bb2, hcat + 128, mT, 64, 256, 64, 256, 1);

    // layer 3: concat(h0,h1,h2)=hcat[N,256] -> nf[N,256]
    hipMemsetAsync(agg256, 0, (size_t)N * 256 * 4, stream);
    {
        long long tot = (long long)E * 64;
        scatter_add_kernel<<<(tot + 255) / 256, 256, 0, stream>>>(hcat, 256, srcI, dstI, agg256, 256, 64, tot);
    }
    wmma_gemm_fused<<<dim3(gxN, 256 / 64), gblk, 0, stream>>>(
        hcat, ws3p, agg256, wn3p, deg, bb3, nf, mT, 256, 256, 256, 256, 1);

    // pooling
    hipMemsetAsync(gmaxU, 0, (size_t)G * 256 * 4, stream);
    hipMemsetAsync(gsum, 0, (size_t)G * 256 * 4, stream);
    hipMemsetAsync(gcnt, 0, (size_t)G * 4, stream);
    pool_kernel<<<((size_t)N * 64 + 255) / 256, 256, 0, stream>>>(nf, bat, gmaxU, gsum, gcnt, N);
    pool_finalize<<<((size_t)G * 512 + 255) / 256, 256, 0, stream>>>(gmaxU, gsum, gcnt, gvec, G);

    // dense head: [G,512] -> leaky -> [G,512] -> [G,256]
    wmma_gemm_fused<<<dim3(gxG, 512 / 64), gblk, 0, stream>>>(
        gvec, wd1p, nullptr, nullptr, nullptr, bd1, gtmp, mTg, 512, 512, 0, 512, 1);
    wmma_gemm_fused<<<dim3(gxG, 256 / 64), gblk, 0, stream>>>(
        gtmp, wd2p, nullptr, nullptr, nullptr, bd2, (float*)d_out, mTg, 512, 512, 0, 256, 0);
}
